// ReferenceGroupLimitedRouter_7670811591157
// MI455X (gfx1250) — matrix operand order repaired and checked by reading, not compiled, not run on hardware
//
#include <hip/hip_runtime.h>
#include <hip/hip_bf16.h>
#include <stdint.h>

#define NUM_EXPERTS 256
#define HID 7168
#define B_TOKENS 16384
#define TOP_K 8
#define N_GROUP 8
#define TOPK_GROUP 4
#define ROUTED_SCALE 2.5f

typedef __attribute__((ext_vector_type(16))) __bf16 v16bf;
typedef __attribute__((ext_vector_type(8)))  __bf16 v8bf;
typedef __attribute__((ext_vector_type(8)))  float  v8f;

typedef unsigned int uint32x4 __attribute__((ext_vector_type(4)));
typedef int          int32x8  __attribute__((ext_vector_type(8)));
typedef int          int32x4  __attribute__((ext_vector_type(4)));

// LDS layout written by the TDM: per array, 256 expert rows of 32 bf16 (64B)
// padded by 16B -> 80B row stride (16B aligned, bank-conflict free reads).
#define ROW_BF16   40            // 80 bytes / 2
#define ARR_BYTES  (256 * 80)    // 20480 B per (hi|lo) array
#define BUF_BYTES  (2 * ARR_BYTES)

// ---------------------------------------------------------------------------
// Kernel 1: split f32 weight into bf16 hi/lo pair (bf16x3 decomposition).
// ---------------------------------------------------------------------------
__global__ void __launch_bounds__(256)
convert_weight_kernel(const float* __restrict__ w,
                      __bf16* __restrict__ whi,
                      __bf16* __restrict__ wlo) {
    size_t i = (size_t)blockIdx.x * blockDim.x + threadIdx.x;
    float x = w[i];
    __bf16 h = (__bf16)x;
    whi[i] = h;
    wlo[i] = (__bf16)(x - (float)h);
}

// ---------------------------------------------------------------------------
// TDM fill: one 2D tile (256 experts x 32 K, bf16) global -> LDS.
// D# per CDNA5 ISA 8.3/8.4. data_size=1 (2B). Pad 16B after every 64B so the
// LDS row stride is 80B. Group2/3 zero (2D tensor, tile_dim2..4 unused).
// This toolchain exposes the 6-arg builtin (extra int32x8 group before cpol).
// ---------------------------------------------------------------------------
__device__ __forceinline__ void tdm_fill_tile(unsigned lds_addr, const __bf16* gsrc) {
    unsigned long long ga = (unsigned long long)(size_t)gsrc;
    uint32x4 g0;
    g0[0] = 1u;                                        // count=1, user mode
    g0[1] = lds_addr;                                  // lds_addr (bytes)
    g0[2] = (unsigned)(ga & 0xFFFFFFFFu);              // global_addr[31:0]
    g0[3] = (unsigned)((ga >> 32) & 0x1FFFFFFu)        // global_addr[56:32]
          | (2u << 30);                                // type=2 ("image")
    int32x8 g1;
    g1[0] = (int)((1u << 16)      // data_size = 1 -> 2 bytes
                | (1u << 20)      // pad_enable
                | (3u << 22)      // pad_interval: 3 -> 16 DWORDs = 64B
                | (3u << 25));    // pad_amount:   3 -> 4 DWORDs = 16B
    g1[1] = (int)((unsigned)HID << 16);                         // tensor_dim0 lo16
    g1[2] = (int)(((unsigned)HID >> 16) | (256u << 16));        // dim0 hi16 | dim1 lo16
    g1[3] = (int)((256u >> 16) | (32u << 16));                  // dim1 hi16 | tile_dim0=32
    g1[4] = (int)256u;                                          // tile_dim1=256 (tile_dim2=0)
    g1[5] = (int)HID;                                           // tensor_dim0_stride lo32
    g1[6] = 0;                                                  // stride hi16 | dim1_stride lo16
    g1[7] = 0;
    int32x4 gz4 = {0, 0, 0, 0};
    int32x8 gz8 = {0, 0, 0, 0, 0, 0, 0, 0};
    __builtin_amdgcn_tensor_load_to_lds(g0, g1, gz4, gz4, gz8, 0);
}

// ---------------------------------------------------------------------------
// Kernel 2: router logits GEMM via v_wmma_f32_16x16x32_bf16 (bf16x3).
// Block = 256 threads = 8 waves. Wave w owns rows [blk*128 + 16w, +16) x all
// 256 experts. B (hi/lo weight tiles) staged in LDS by the TDM, double
// buffered; A built in-register from f32 hidden each K-step. B reads are
// software-pipelined one tile ahead (2 register sets) so the ds_load latency
// hides behind the 3-WMMA chain of the previous tile.
// ---------------------------------------------------------------------------
__global__ void __launch_bounds__(256)
router_gemm_kernel(const float* __restrict__ hidden,
                   const __bf16* __restrict__ whi,
                   const __bf16* __restrict__ wlo,
                   float* __restrict__ logits) {
    __shared__ __align__(128) __bf16 smem[2][2][256 * ROW_BF16]; // 80 KB

    const int lane = threadIdx.x & 31;
    const int wave = threadIdx.x >> 5;
    const int half = lane >> 4;     // 0: lanes 0-15, 1: lanes 16-31
    const int mrow = lane & 15;
    const int row_base = (blockIdx.x * 8 + wave) * 16;

    // Generic shared address = {shared aperture, lds_offset}: low 32 bits are
    // the LDS byte offset needed by the TDM descriptor.
    const unsigned lds_base = (unsigned)(size_t)&smem[0][0][0];

    v8f acc[16] = {};

    const float* arow = hidden + (size_t)(row_base + mrow) * HID;

    // Prologue: fill buffer 0 (hi then lo) from wave 0 only.
    if (wave == 0) {
        tdm_fill_tile(lds_base, whi);
        tdm_fill_tile(lds_base + ARR_BYTES, wlo);
    }

    union Bt { v16bf v; v8bf h[2]; };

    int cur = 0;
    for (int k0 = 0; k0 < HID; k0 += 32) {
        if (wave == 0) {
            if (k0 + 32 < HID) {
                const unsigned nb = lds_base + (unsigned)(cur ^ 1) * BUF_BYTES;
                tdm_fill_tile(nb, whi + k0 + 32);
                tdm_fill_tile(nb + ARR_BYTES, wlo + k0 + 32);
                __builtin_amdgcn_s_wait_tensorcnt(2);   // fill(k0) complete
            } else {
                __builtin_amdgcn_s_wait_tensorcnt(0);
            }
        }
        __syncthreads();   // buffer `cur` ready for everyone

        // Build A (16x32 bf16, hi/lo) matching B: VGPR v holds
        // K = 16*half + 2*v + {0,1}
        v16bf a_hi, a_lo;
        #pragma unroll
        for (int v = 0; v < 8; ++v) {
            const int kk = half * 16 + v * 2;
            float2 x = *(const float2*)(arow + k0 + kk);
            __bf16 h0 = (__bf16)x.x, h1 = (__bf16)x.y;
            a_hi[2 * v]     = h0;
            a_hi[2 * v + 1] = h1;
            a_lo[2 * v]     = (__bf16)(x.x - (float)h0);
            a_lo[2 * v + 1] = (__bf16)(x.y - (float)h1);
        }

        // B (32x16 bf16): lane = column N = t*16 + mrow, 16 K-consecutive
        // values at K offset half*16; LDS row stride 80B (16B aligned).
        Bt bhv[2], blv[2];
        {
            const __bf16* ph = &smem[cur][0][mrow * ROW_BF16 + half * 16];
            const __bf16* pl = &smem[cur][1][mrow * ROW_BF16 + half * 16];
            bhv[0].h[0] = *(const v8bf*)ph;  bhv[0].h[1] = *(const v8bf*)(ph + 8);
            blv[0].h[0] = *(const v8bf*)pl;  blv[0].h[1] = *(const v8bf*)(pl + 8);
        }
        #pragma unroll
        for (int t = 0; t < 16; ++t) {
            if (t < 15) {   // prefetch tile t+1 into the alternate set
                const int e = (t + 1) * 16 + mrow;
                const __bf16* ph = &smem[cur][0][e * ROW_BF16 + half * 16];
                const __bf16* pl = &smem[cur][1][e * ROW_BF16 + half * 16];
                bhv[(t + 1) & 1].h[0] = *(const v8bf*)ph;
                bhv[(t + 1) & 1].h[1] = *(const v8bf*)(ph + 8);
                blv[(t + 1) & 1].h[0] = *(const v8bf*)pl;
                blv[(t + 1) & 1].h[1] = *(const v8bf*)(pl + 8);
            }
            const int c = t & 1;
            acc[t] = __builtin_amdgcn_wmma_f32_16x16x32_bf16(
                false, a_hi, false, bhv[c].v, (short)0, acc[t], false, false);
            acc[t] = __builtin_amdgcn_wmma_f32_16x16x32_bf16(
                false, a_hi, false, blv[c].v, (short)0, acc[t], false, false);
            acc[t] = __builtin_amdgcn_wmma_f32_16x16x32_bf16(
                false, a_lo, false, bhv[c].v, (short)0, acc[t], false, false);
        }
        __syncthreads();   // all waves done reading `cur` before its refill
        cur ^= 1;
    }

    // C/D layout: VGPR r, lanes 0-15 -> M=r, lanes 16-31 -> M=r+8; N = lane%16.
    #pragma unroll
    for (int t = 0; t < 16; ++t) {
        #pragma unroll
        for (int r = 0; r < 8; ++r) {
            const int row = row_base + r + 8 * half;
            logits[(size_t)row * NUM_EXPERTS + t * 16 + mrow] = acc[t][r];
        }
    }
}

// ---------------------------------------------------------------------------
// Kernel 3: sigmoid + group-limited top-k routing. 1 wave / 32 tokens.
// LDS stride 257 (257 % 64 == 1) -> conflict-free per-lane serial scans.
// Tie-breaking matches jax.lax.top_k: strict '>' over ascending index.
// ---------------------------------------------------------------------------
#define TOK_PER_BLK 32
#define LDS_STRIDE  257

__global__ void __launch_bounds__(32)
routing_topk_kernel(const float* __restrict__ logits,
                    const float* __restrict__ bias,
                    int* __restrict__ out_idx,
                    float* __restrict__ out_w) {
    __shared__ float sc[TOK_PER_BLK][LDS_STRIDE];
    __shared__ float bias_s[NUM_EXPERTS];

    const int tid = threadIdx.x;
    const int tok_base = blockIdx.x * TOK_PER_BLK;

    for (int e = tid; e < NUM_EXPERTS; e += 32) bias_s[e] = bias[e];

    const float* src = logits + (size_t)tok_base * NUM_EXPERTS;
    for (int i = tid; i < TOK_PER_BLK * NUM_EXPERTS; i += 32) {
        float x = src[i];
        sc[i >> 8][i & 255] = 1.0f / (1.0f + expf(-x));
    }
    __syncthreads();

    const int t = tid;  // one token per lane

    float gscore[N_GROUP];
    #pragma unroll
    for (int g = 0; g < N_GROUP; ++g) {
        float m1 = -1e30f, m2 = -1e30f;
        for (int e = g * 32; e < g * 32 + 32; ++e) {
            float v = sc[t][e] + bias_s[e];
            if (v > m1)      { m2 = m1; m1 = v; }
            else if (v > m2) { m2 = v; }
        }
        gscore[g] = m1 + m2;
    }

    unsigned gm = 0;
    #pragma unroll
    for (int i = 0; i < TOPK_GROUP; ++i) {
        int best = 0; float bv = -1e30f;
        #pragma unroll
        for (int g = 0; g < N_GROUP; ++g) {
            if ((gm >> g) & 1) continue;
            if (gscore[g] > bv) { bv = gscore[g]; best = g; }
        }
        gm |= 1u << best;
    }

    unsigned picked[8] = {0, 0, 0, 0, 0, 0, 0, 0};
    int   idx[TOP_K];
    float wv[TOP_K];
    float wsum = 0.0f;
    #pragma unroll
    for (int i = 0; i < TOP_K; ++i) {
        int best = 0; float bv = -1e30f;
        for (int e = 0; e < NUM_EXPERTS; ++e) {
            if ((picked[e >> 5] >> (e & 31)) & 1) continue;
            float v = ((gm >> (e >> 5)) & 1) ? (sc[t][e] + bias_s[e]) : 0.0f;
            if (v > bv) { bv = v; best = e; }
        }
        picked[best >> 5] |= 1u << (best & 31);
        idx[i] = best;
        float w = sc[t][best];   // weights use UN-biased sigmoid scores
        wv[i] = w;
        wsum += w;
    }

    const float inv = ROUTED_SCALE / wsum;
    const int token = tok_base + t;
    #pragma unroll
    for (int i = 0; i < TOP_K; ++i) {
        out_idx[token * TOP_K + i] = idx[i];
        out_w[token * TOP_K + i]   = wv[i] * inv;
    }
}

// ---------------------------------------------------------------------------
extern "C" void kernel_launch(void* const* d_in, const int* in_sizes, int n_in,
                              void* d_out, int out_size, void* d_ws, size_t ws_size,
                              hipStream_t stream) {
    const float* hidden = (const float*)d_in[0];   // [16384, 7168] f32
    const float* weight = (const float*)d_in[1];   // [256, 7168]   f32
    const float* bias   = (const float*)d_in[2];   // [256]         f32

    const size_t WN = (size_t)NUM_EXPERTS * HID;   // 1,835,008
    __bf16* whi = (__bf16*)d_ws;                               // 3.67 MB
    __bf16* wlo = whi + WN;                                    // 3.67 MB
    float*  logits = (float*)((char*)d_ws + 2 * WN * sizeof(__bf16)); // 16.8 MB

    int*   out_idx = (int*)d_out;                               // [16384, 8] i32
    float* out_w   = (float*)((int*)d_out + (size_t)B_TOKENS * TOP_K); // [16384, 8] f32

    convert_weight_kernel<<<(int)(WN / 256), 256, 0, stream>>>(weight, whi, wlo);
    router_gemm_kernel<<<B_TOKENS / 128, 256, 0, stream>>>(hidden, whi, wlo, logits);
    routing_topk_kernel<<<B_TOKENS / TOK_PER_BLK, 32, 0, stream>>>(logits, bias, out_idx, out_w);
}
